// PairAttnDecoderRNN_74517682586371
// MI455X (gfx1250) — matrix operand order repaired and checked by reading, not compiled, not run on hardware
//
#include <hip/hip_runtime.h>

typedef __attribute__((ext_vector_type(16))) __bf16 v16bf;
typedef __attribute__((ext_vector_type(8)))  float  v8f;

#define B_      256
#define H_      1024
#define M_      20
#define KTOP_   20
#define VOCAB_  32000

#define LSTR    40   // LDS row stride in ushorts (80 B: keeps 16-B alignment, spreads banks)

// ---------- f32 -> packed bf16 pair ----------
// Plain fptrunc: clang emits the hardware cvt (v_cvt_pk_bf16_f32 class) when
// the target supports it, and a correct RNE software sequence otherwise.
static __device__ __forceinline__ unsigned pack2(float lo, float hi) {
  union { __bf16 h[2]; unsigned u; } c;
  c.h[0] = (__bf16)lo;
  c.h[1] = (__bf16)hi;
  return c.u;
}

union FragU {
  uint4  q[2];
  v16bf  v;
};

// =====================================================================
// Generic bf16-WMMA GEMM:  C[M,N] = op(A) * op(B) (+ bias)
//   A: row-major M x K (fp32), optional per-row gather via gidx
//   B: TRANSB==0 -> row-major K x N ; TRANSB==1 -> row-major N x K (B^T)
// Block tile 64x128, k-step 32, 256 threads (8 wave32), 4 WMMA / wave / k
// All LDS traffic is 16-B aligned ds_load_b128 / packed stores.
// Requires: M % 64 == 0, N % 128 == 0, K % 32 == 0 (true for all calls)
// =====================================================================
template<bool TRANSB>
__global__ __launch_bounds__(256)
void gemm_bf16_wmma(const float* __restrict__ A, const float* __restrict__ Bm,
                    const float* __restrict__ bias, const int* __restrict__ gidx,
                    float* __restrict__ C, int M, int N, int K)
{
  // lA[row][k] : 64 x 32 bf16, stride 40 ushorts (80 B rows)
  // lB[n][k]   : 128 x 32 bf16 (k-contiguous per column), stride 40 ushorts
  __shared__ __align__(16) unsigned short lA[64 * LSTR];
  __shared__ __align__(16) unsigned short lB[128 * LSTR];

  const int t    = threadIdx.x;
  const int lane = t & 31;
  const int wave = t >> 5;
  const int wm   = wave & 3;        // 4 waves along M (16 rows each)
  const int wn   = wave >> 2;       // 2 waves along N (64 cols each)
  const int half = lane >> 4;       // 0: lanes 0-15, 1: lanes 16-31
  const int lr   = lane & 15;
  const int m0   = blockIdx.y * 64;
  const int n0   = blockIdx.x * 128;

  v8f acc[4] = {};

  // ---- A staging assignment: row 0..63, 8 consecutive k ----
  const int  arow_l = t >> 2;
  const int  acol0  = (t & 3) * 8;
  long arow_g = m0 + arow_l;
  if (gidx) arow_g = gidx[arow_g];
  const float* Aptr = A + (size_t)arow_g * (size_t)K;

  // ---- B staging assignment ----
  // !TRANSB: thread packs k-pair (2*kp, 2*kp+1) x 8 n values
  const int kp = t >> 4;            // 0..15
  const int nc = (t & 15) * 8;      // n offset 0..120
  // TRANSB: thread handles one n row, 16 consecutive k
  const int nl  = t >> 1;           // 0..127
  const int kb2 = (t & 1) * 16;     // 0 or 16

  for (int kk = 0; kk < K; kk += 32) {
    // ================= stage A (one b128 store per thread) =================
    {
      const float4 f0 = *(const float4*)(Aptr + kk + acol0);
      const float4 f1 = *(const float4*)(Aptr + kk + acol0 + 4);
      uint4 p;
      p.x = pack2(f0.x, f0.y); p.y = pack2(f0.z, f0.w);
      p.z = pack2(f1.x, f1.y); p.w = pack2(f1.z, f1.w);
      *(uint4*)&lA[arow_l * LSTR + acol0] = p;
      if (kk + 32 < K)
        __builtin_prefetch(Aptr + kk + 32 + acol0, 0, 1);
    }

    // ================= stage B into lB[n][k] =================
    if (!TRANSB) {
      const float* r0 = Bm + (size_t)(kk + 2 * kp) * (size_t)N + n0 + nc;
      const float* r1 = r0 + N;
      const float4 a0 = *(const float4*)(r0);
      const float4 a1 = *(const float4*)(r0 + 4);
      const float4 b0 = *(const float4*)(r1);
      const float4 b1 = *(const float4*)(r1 + 4);
      // transpose-on-store: one packed dword per n (k-pair contiguous)
      *(unsigned*)&lB[(nc + 0) * LSTR + 2 * kp] = pack2(a0.x, b0.x);
      *(unsigned*)&lB[(nc + 1) * LSTR + 2 * kp] = pack2(a0.y, b0.y);
      *(unsigned*)&lB[(nc + 2) * LSTR + 2 * kp] = pack2(a0.z, b0.z);
      *(unsigned*)&lB[(nc + 3) * LSTR + 2 * kp] = pack2(a0.w, b0.w);
      *(unsigned*)&lB[(nc + 4) * LSTR + 2 * kp] = pack2(a1.x, b1.x);
      *(unsigned*)&lB[(nc + 5) * LSTR + 2 * kp] = pack2(a1.y, b1.y);
      *(unsigned*)&lB[(nc + 6) * LSTR + 2 * kp] = pack2(a1.z, b1.z);
      *(unsigned*)&lB[(nc + 7) * LSTR + 2 * kp] = pack2(a1.w, b1.w);
      if (kk + 32 < K)
        __builtin_prefetch(r0 + 32ll * (long)N, 0, 1);   // next k-tile, same n
    } else {
      const float* bp = Bm + (size_t)(n0 + nl) * (size_t)K + kk + kb2;
      const float4 f0 = *(const float4*)(bp);
      const float4 f1 = *(const float4*)(bp + 4);
      const float4 f2 = *(const float4*)(bp + 8);
      const float4 f3 = *(const float4*)(bp + 12);
      uint4 p0, p1;
      p0.x = pack2(f0.x, f0.y); p0.y = pack2(f0.z, f0.w);
      p0.z = pack2(f1.x, f1.y); p0.w = pack2(f1.z, f1.w);
      p1.x = pack2(f2.x, f2.y); p1.y = pack2(f2.z, f2.w);
      p1.z = pack2(f3.x, f3.y); p1.w = pack2(f3.z, f3.w);
      *(uint4*)&lB[nl * LSTR + kb2]     = p0;
      *(uint4*)&lB[nl * LSTR + kb2 + 8] = p1;
      if (kk + 32 < K)
        __builtin_prefetch(bp + 32, 0, 1);
    }
    __syncthreads();

    // ---- A fragment: two b128 LDS loads ----
    // lane elements: k = {kb..kb+7, kb+16..kb+23}, kb = half*8
    FragU af;
    {
      const int ar = wm * 16 + lr;
      const int kb = half * 8;
      af.q[0] = *(const uint4*)&lA[ar * LSTR + kb];
      af.q[1] = *(const uint4*)&lA[ar * LSTR + kb + 16];
    }

    // ---- 4 N-subtiles: each B fragment is two b128 LDS loads ----
    #pragma unroll
    for (int nt = 0; nt < 4; ++nt) {
      FragU bf;
      const int c  = wn * 64 + nt * 16 + lr;
      const int kb = half * 8;           // lane elements: k = {kb..kb+7, kb+16..kb+23}
      bf.q[0] = *(const uint4*)&lB[c * LSTR + kb];
      bf.q[1] = *(const uint4*)&lB[c * LSTR + kb + 16];
      acc[nt] = __builtin_amdgcn_wmma_f32_16x16x32_bf16(
          false, af.v, false, bf.v, (short)0, acc[nt], false, false);
    }
    __syncthreads();
  }

  // ---- epilogue: C/D layout (lane<16 -> M=v, lane>=16 -> M=v+8) ----
  #pragma unroll
  for (int nt = 0; nt < 4; ++nt) {
    const int c = n0 + wn * 64 + nt * 16 + lr;
    const float bv = bias ? bias[c] : 0.0f;
    #pragma unroll
    for (int v = 0; v < 8; ++v) {
      const int r = m0 + wm * 16 + v + half * 8;
      C[(size_t)r * (size_t)N + c] = acc[nt][v] + bv;
    }
  }
}

// =====================================================================
// emb gather: out_pair[b, 0:H] = emb_dec[input_ids[b]]
// =====================================================================
__global__ __launch_bounds__(256)
void emb_gather(const int* __restrict__ ids, const float* __restrict__ emb_dec,
                float* __restrict__ out_pair)
{
  const int b = blockIdx.x;
  const size_t src = (size_t)ids[b] * H_;
  for (int d = threadIdx.x; d < H_; d += 256)
    out_pair[(size_t)b * (3 * H_) + d] = emb_dec[src + d];
}

// =====================================================================
// Ingredient attention: scores -> argmax (softmax is monotone) -> comp_ids
// =====================================================================
__global__ __launch_bounds__(256)
void ingr_attn(const float* __restrict__ query, const float* __restrict__ key,
               const float* __restrict__ wa, const int* __restrict__ input_tensor,
               const int* __restrict__ pairing_ids, int* __restrict__ comp_ids)
{
  __shared__ float red[256];
  __shared__ float sc[M_];
  __shared__ int   s_ingr;
  const int b = blockIdx.x, t = threadIdx.x;

  for (int m = 0; m < M_; ++m) {
    float p = 0.f;
    const float* kr = key + (size_t)(b * M_ + m) * H_;
    for (int h = t; h < H_; h += 256)
      p += tanhf(query[(size_t)b * H_ + h] + kr[h]) * wa[h];
    red[t] = p; __syncthreads();
    for (int s = 128; s > 0; s >>= 1) {
      if (t < s) red[t] += red[t + s];
      __syncthreads();
    }
    if (t == 0) sc[m] = red[0];
    __syncthreads();
  }
  if (t == 0) {
    int arg = 0; float best = sc[0];
    for (int m = 1; m < M_; ++m)
      if (sc[m] > best) { best = sc[m]; arg = m; }   // first-max tie-break
    s_ingr = input_tensor[b * M_ + arg];
  }
  __syncthreads();
  if (t < KTOP_) comp_ids[b * KTOP_ + t] = pairing_ids[s_ingr * KTOP_ + t];
}

// =====================================================================
// Pairing attention: s_p, softmax*raw weights, weighted sum of comp_emb
// writes out_pair[b, H:3H]
// =====================================================================
__global__ __launch_bounds__(256)
void pair_attn(const float* __restrict__ qp, const float* __restrict__ kp,
               const float* __restrict__ wa, const float* __restrict__ ba,
               const int* __restrict__ comp_ids, const float* __restrict__ enc_emb,
               float* __restrict__ out_pair)
{
  __shared__ float red[256];
  __shared__ float sp[KTOP_];
  __shared__ float aw[KTOP_];
  const int b = blockIdx.x, t = threadIdx.x;

  for (int k = 0; k < KTOP_; ++k) {
    float p = 0.f;
    const float* kr = kp + (size_t)(b * KTOP_ + k) * H_;
    for (int h = t; h < H_; h += 256)
      p += tanhf(qp[(size_t)b * H_ + h] + kr[h]) * wa[h];
    red[t] = p; __syncthreads();
    for (int s = 128; s > 0; s >>= 1) {
      if (t < s) red[t] += red[t + s];
      __syncthreads();
    }
    if (t == 0) sp[k] = red[0] + ba[0];
    __syncthreads();
  }
  if (t == 0) {
    float mx = sp[0];
    for (int k = 1; k < KTOP_; ++k) mx = fmaxf(mx, sp[k]);
    float sum = 0.f;
    for (int k = 0; k < KTOP_; ++k) { float e = expf(sp[k] - mx); aw[k] = e; sum += e; }
    const float inv = 1.f / sum;
    for (int k = 0; k < KTOP_; ++k) aw[k] = aw[k] * inv * sp[k];  // a_p * s_p
  }
  __syncthreads();
  for (int d = t; d < 2 * H_; d += 256) {
    float a = 0.f;
    #pragma unroll
    for (int k = 0; k < KTOP_; ++k)
      a += aw[k] * enc_emb[(size_t)comp_ids[b * KTOP_ + k] * (2 * H_) + d];
    out_pair[(size_t)b * (3 * H_) + H_ + d] = a;
  }
}

// =====================================================================
// GRU step (torch gate order r,z,n): h_new = (1-z)*n + z*h
// =====================================================================
__global__ __launch_bounds__(256)
void gru_step(const float* __restrict__ gi, const float* __restrict__ gh,
              const float* __restrict__ h, float* __restrict__ hnew)
{
  const int i = blockIdx.x * 256 + threadIdx.x;   // < B_*H_
  const int b = i >> 10, j = i & (H_ - 1);
  const float* gib = gi + (size_t)b * (3 * H_);
  const float* ghb = gh + (size_t)b * (3 * H_);
  const float r = 1.f / (1.f + expf(-(gib[j]          + ghb[j])));
  const float z = 1.f / (1.f + expf(-(gib[H_ + j]     + ghb[H_ + j])));
  const float n = tanhf(gib[2 * H_ + j] + r * ghb[2 * H_ + j]);
  hnew[i] = (1.f - z) * n + z * h[i];
}

// =====================================================================
extern "C" void kernel_launch(void* const* d_in, const int* in_sizes, int n_in,
                              void* d_out, int out_size, void* d_ws, size_t ws_size,
                              hipStream_t stream)
{
  const int*   input_ids    = (const int*)  d_in[0];
  const float* hidden       = (const float*)d_in[1];   // (B,H) flat
  const float* enc_out      = (const float*)d_in[2];   // (M,B,2H) == flat (B*M,2H) after reshape
  const int*   input_tensor = (const int*)  d_in[3];
  const int*   pairing_ids  = (const int*)  d_in[4];
  const float* emb_dec      = (const float*)d_in[5];
  const float* enc_emb      = (const float*)d_in[6];
  const float* Wq_i  = (const float*)d_in[7];
  const float* Wk_i  = (const float*)d_in[8];
  const float* wa_i  = (const float*)d_in[9];
  const float* Wq_p  = (const float*)d_in[10];
  const float* Wk_p  = (const float*)d_in[11];
  const float* wa_p  = (const float*)d_in[12];
  const float* ba_p  = (const float*)d_in[13];
  const float* W_lin = (const float*)d_in[14];
  const float* b_lin = (const float*)d_in[15];
  const float* W_ih  = (const float*)d_in[16];
  const float* W_hh  = (const float*)d_in[17];
  const float* b_ih  = (const float*)d_in[18];
  const float* b_hh  = (const float*)d_in[19];
  const float* W_out = (const float*)d_in[20];
  const float* b_out = (const float*)d_in[21];
  float* logits = (float*)d_out;

  // ---- workspace carve-up (~36 MB) ----
  float* ws       = (float*)d_ws;
  float* out_pair = ws;                              // B x 3H
  float* query    = out_pair + (size_t)B_ * 3 * H_;  // B x H
  float* qp       = query    + (size_t)B_ * H_;      // B x H
  float* keybuf   = qp       + (size_t)B_ * H_;      // (B*M) x H  (reused for k_p)
  float* xbuf     = keybuf   + (size_t)B_ * M_ * H_; // B x 2H
  float* gi       = xbuf     + (size_t)B_ * 2 * H_;  // B x 3H
  float* gh       = gi       + (size_t)B_ * 3 * H_;  // B x 3H
  float* hnew     = gh       + (size_t)B_ * 3 * H_;  // B x H
  int*   comp_ids = (int*)(hnew + (size_t)B_ * H_);  // B*K ints

  const dim3 blk(256);

  // 1) emb = emb_dec[input_ids]  -> out_pair[:, :H]
  emb_gather<<<B_, blk, 0, stream>>>(input_ids, emb_dec, out_pair);

  // 2) query = h @ Wq_i ; q_p = h @ Wq_p      (256x1024 @ 1024x1024)
  gemm_bf16_wmma<false><<<dim3(H_ / 128, B_ / 64), blk, 0, stream>>>(
      hidden, Wq_i, nullptr, nullptr, query, B_, H_, H_);
  gemm_bf16_wmma<false><<<dim3(H_ / 128, B_ / 64), blk, 0, stream>>>(
      hidden, Wq_p, nullptr, nullptr, qp, B_, H_, H_);

  // 3) key = enc_flat @ Wk_i                  (5120x2048 @ 2048x1024)
  gemm_bf16_wmma<false><<<dim3(H_ / 128, (B_ * M_) / 64), blk, 0, stream>>>(
      enc_out, Wk_i, nullptr, nullptr, keybuf, B_ * M_, H_, 2 * H_);

  // 4) ingredient attention -> comp_ids
  ingr_attn<<<B_, blk, 0, stream>>>(query, keybuf, wa_i, input_tensor,
                                    pairing_ids, comp_ids);

  // 5) k_p = enc_emb[comp_ids] @ Wk_p         (gathered-A GEMM, reuses keybuf)
  gemm_bf16_wmma<false><<<dim3(H_ / 128, (B_ * KTOP_) / 64), blk, 0, stream>>>(
      enc_emb, Wk_p, nullptr, comp_ids, keybuf, B_ * KTOP_, H_, 2 * H_);

  // 6) pairing attention -> out_pair[:, H:3H]
  pair_attn<<<B_, blk, 0, stream>>>(qp, keybuf, wa_p, ba_p, comp_ids,
                                    enc_emb, out_pair);

  // 7) x = out_pair @ W_lin + b_lin           (256x3072 @ 3072x2048)
  gemm_bf16_wmma<false><<<dim3((2 * H_) / 128, B_ / 64), blk, 0, stream>>>(
      out_pair, W_lin, b_lin, nullptr, xbuf, B_, 2 * H_, 3 * H_);

  // 8) gi = x @ W_ih^T + b_ih                 (256x2048 @ (3072x2048)^T)
  gemm_bf16_wmma<true><<<dim3((3 * H_) / 128, B_ / 64), blk, 0, stream>>>(
      xbuf, W_ih, b_ih, nullptr, gi, B_, 3 * H_, 2 * H_);

  // 9) gh = h @ W_hh^T + b_hh                 (256x1024 @ (3072x1024)^T)
  gemm_bf16_wmma<true><<<dim3((3 * H_) / 128, B_ / 64), blk, 0, stream>>>(
      hidden, W_hh, b_hh, nullptr, gh, B_, 3 * H_, H_);

  // 10) GRU elementwise
  gru_step<<<(B_ * H_) / 256, blk, 0, stream>>>(gi, gh, hidden, hnew);

  // 11) logits = h_new @ W_out + b_out        (256x1024 @ 1024x32000)
  gemm_bf16_wmma<false><<<dim3(VOCAB_ / 128, B_ / 64), blk, 0, stream>>>(
      hnew, W_out, b_out, nullptr, logits, B_, VOCAB_, H_);
}
